// GAT_Net_4209067950743
// MI455X (gfx1250) — matrix operand order repaired and checked by reading, not compiled, not run on hardware
//
#include <hip/hip_runtime.h>
#include <hip/hip_bf16.h>
#include <math.h>

// ---------------------------------------------------------------------------
// GAT 2-layer forward for MI455X (gfx1250, wave32).
//  - GEMMs via v_wmma_f32_16x16x32_f16 (f16 operands, f32 accumulate).
//  - Edge softmax/aggregation via f32 global atomics (L2-resident working set).
// Problem dims: N=50000 nodes, E=800000 edges, F=256, H1=8, C1=32, C2=32.
// ---------------------------------------------------------------------------

typedef __attribute__((ext_vector_type(16))) _Float16 v16h;
typedef __attribute__((ext_vector_type(8)))  _Float16 v8h;
typedef __attribute__((ext_vector_type(8)))  float    v8f;

// ------------------------- small helpers -----------------------------------

__device__ __forceinline__ float lrelu02(float x) {
    return x > 0.0f ? x : 0.2f * x;
}

__device__ __forceinline__ void atomicMaxF(float* addr, float v) {
    // Sign-aware integer punning: valid because IEEE754 floats compare like
    // ints when >=0 and like reversed uints when <0.  Init value must be -inf.
    if (v >= 0.0f) {
        atomicMax((int*)addr, __float_as_int(v));
    } else {
        atomicMin((unsigned int*)addr, (unsigned int)__float_as_int(v));
    }
}

__global__ void fill_f32(float* __restrict__ p, float v, long long n) {
    long long i = (long long)blockIdx.x * blockDim.x + threadIdx.x;
    if (i < n) p[i] = v;
}

__global__ void cvt_f32_to_f16(const float* __restrict__ src,
                               _Float16* __restrict__ dst, long long n) {
    long long i = (long long)blockIdx.x * blockDim.x + threadIdx.x;
    if (i < n) dst[i] = (_Float16)src[i];
}

// W: [R][Ccols] f32 -> WT: [Ccols][R] f16   (so B columns become contiguous)
__global__ void transpose_to_f16(const float* __restrict__ W,
                                 _Float16* __restrict__ WT, int R, int Ccols) {
    int t = blockIdx.x * blockDim.x + threadIdx.x;
    if (t >= R * Ccols) return;
    int r = t / Ccols, c = t % Ccols;
    WT[(long long)c * R + r] = (_Float16)W[(long long)r * Ccols + c];
}

// ------------------------- WMMA GEMM ----------------------------------------
// C[M][N] (f32) = A[M][K] (f16, row major) * BT[N][K] (f16, i.e. B transposed)
// One wave computes a 16 x (16*NT) tile.  blockDim = 128 (4 waves).
// Fragment layout per CDNA5 ISA 7.12.2 (wave32):
//   A: lane l (m=l%16, h=l/16) holds K = [8h,8h+8) and [16+8h,24+8h)
//   B: lane l (n=l%16, h=l/16) holds K = [8h,8h+8) and [16+8h,24+8h) (same as A)
//   C: lane l (n=l%16) element r holds row m = r + 8h
template <int NT>
__global__ void gemm_f16_wmma(const _Float16* __restrict__ A,
                              const _Float16* __restrict__ BT,
                              float* __restrict__ C, int M, int N, int K) {
    const int lane = threadIdx.x & 31;
    const int wid  = threadIdx.x >> 5;
    const int mt   = blockIdx.x * (blockDim.x >> 5) + wid;
    const int Mt   = M >> 4;
    if (mt >= Mt) return;                      // wave-uniform: EXEC stays full

    const int half = lane >> 4;
    const int l16  = lane & 15;
    const int n0   = blockIdx.y * (16 * NT);
    const _Float16* arow = A + (long long)((mt << 4) + l16) * K;

    v8f acc[NT] = {};

    for (int k = 0; k < K; k += 32) {
        v16h a;
        *((v8h*)&a)     = *(const v8h*)(arow + k + 8 * half);
        *((v8h*)&a + 1) = *(const v8h*)(arow + k + 16 + 8 * half);
#pragma unroll
        for (int i = 0; i < NT; ++i) {
            const int ncol = n0 + i * 16 + l16;
            v16h b;
            *((v8h*)&b)     = *(const v8h*)(BT + (long long)ncol * K + k + 8 * half);
            *((v8h*)&b + 1) = *(const v8h*)(BT + (long long)ncol * K + k + 16 + 8 * half);
            acc[i] = __builtin_amdgcn_wmma_f32_16x16x32_f16(
                false, a, false, b, (short)0, acc[i], false, false);
        }
    }

#pragma unroll
    for (int i = 0; i < NT; ++i) {
        const int ncol = n0 + i * 16 + l16;
#pragma unroll
        for (int r = 0; r < 8; ++r) {
            const int m = (mt << 4) + r + 8 * half;
            C[(long long)m * N + ncol] = acc[i][r];
        }
    }
}

// ------------------------- attention pieces ---------------------------------

// alpha_s[n,h] = dot(h[n,h,:], a_src[h,:]); same for a_dst.  One thread per (n,h).
__global__ void alpha_kernel(const float* __restrict__ Hf,
                             const float* __restrict__ a_src,
                             const float* __restrict__ a_dst,
                             float* __restrict__ as_, float* __restrict__ ad_,
                             int Nn, int H, int C) {
    int t = blockIdx.x * blockDim.x + threadIdx.x;
    if (t >= Nn * H) return;
    int n = t / H, h = t % H;
    const float* row = Hf + (long long)n * H * C + h * C;
    float ss = 0.0f, sd = 0.0f;
    for (int c = 0; c < C; ++c) {
        float v = row[c];
        ss += v * a_src[h * C + c];
        sd += v * a_dst[h * C + c];
    }
    as_[t] = ss;
    ad_[t] = sd;
}

// pass 1: segment max of leaky_relu(as[src]+ad[dst]) grouped by dst
__global__ void edge_max_kernel(const int* __restrict__ src,
                                const int* __restrict__ dst,
                                const float* __restrict__ as_,
                                const float* __restrict__ ad_,
                                float* __restrict__ dmax, int E, int H) {
    int t = blockIdx.x * blockDim.x + threadIdx.x;
    if (t >= E * H) return;
    int e = t / H, h = t % H;
    int s = src[e], d = dst[e];
    float v = lrelu02(as_[s * H + h] + ad_[d * H + h]);
    atomicMaxF(&dmax[d * H + h], v);
}

// pass 2: w = exp(e - max[dst]); store per-edge weight; accumulate denominator
__global__ void edge_exp_kernel(const int* __restrict__ src,
                                const int* __restrict__ dst,
                                const float* __restrict__ as_,
                                const float* __restrict__ ad_,
                                const float* __restrict__ dmax,
                                float* __restrict__ ew,
                                float* __restrict__ denom, int E, int H) {
    int t = blockIdx.x * blockDim.x + threadIdx.x;
    if (t >= E * H) return;
    int e = t / H, h = t % H;
    int s = src[e], d = dst[e];
    float v = lrelu02(as_[s * H + h] + ad_[d * H + h]);
    float w = __expf(v - dmax[d * H + h]);
    ew[t] = w;
    atomicAdd(&denom[d * H + h], w);
}

// pass 3: OUT[dst, h, :] += (ew / denom[dst,h]) * X[src, h, :]
__global__ void edge_agg_kernel(const int* __restrict__ src,
                                const int* __restrict__ dst,
                                const float* __restrict__ ew,
                                const float* __restrict__ denom,
                                const float* __restrict__ X,
                                float* __restrict__ OUT, int E, int H, int C) {
    int t = blockIdx.x * blockDim.x + threadIdx.x;
    if (t >= E * H) return;
    int e = t / H, h = t % H;
    int s = src[e], d = dst[e];
    float coeff = ew[t] / (denom[d * H + h] + 1e-16f);
    const int ldx = H * C;
    const float* xrow = X + (long long)s * ldx + h * C;
    float* orow = OUT + (long long)d * ldx + h * C;
    for (int c = 0; c < C; c += 4) {
        float4 v = *(const float4*)(xrow + c);
        atomicAdd(orow + c + 0, coeff * v.x);
        atomicAdd(orow + c + 1, coeff * v.y);
        atomicAdd(orow + c + 2, coeff * v.z);
        atomicAdd(orow + c + 3, coeff * v.w);
    }
}

// layer-1 epilogue: out1 + b1 -> elu -> f16 (feeds layer-2 GEMM)
__global__ void finish1_kernel(const float* __restrict__ acc,
                               const float* __restrict__ b1,
                               _Float16* __restrict__ out_h, long long n, int F) {
    long long i = (long long)blockIdx.x * blockDim.x + threadIdx.x;
    if (i >= n) return;
    float v = acc[i] + b1[i % F];
    v = v > 0.0f ? v : (__expf(v) - 1.0f);
    out_h[i] = (_Float16)v;
}

// layer-2 epilogue: +b2 (mean over H=1 heads is identity), then log_softmax
__global__ void finish2_kernel(const float* __restrict__ acc,
                               const float* __restrict__ b2,
                               float* __restrict__ out, int Nn, int C) {
    int n = blockIdx.x * blockDim.x + threadIdx.x;
    if (n >= Nn) return;
    const float* r = acc + (long long)n * C;
    float v[32];
    float m = -INFINITY;
    for (int c = 0; c < C; ++c) {
        v[c] = r[c] + b2[c];
        m = fmaxf(m, v[c]);
    }
    float s = 0.0f;
    for (int c = 0; c < C; ++c) s += __expf(v[c] - m);
    float lse = m + __logf(s);
    float* orow = out + (long long)n * C;
    for (int c = 0; c < C; ++c) orow[c] = v[c] - lse;
}

// ------------------------- launcher -----------------------------------------

extern "C" void kernel_launch(void* const* d_in, const int* in_sizes, int n_in,
                              void* d_out, int out_size, void* d_ws, size_t ws_size,
                              hipStream_t stream) {
    const float* x     = (const float*)d_in[0];
    const int*   ei    = (const int*)d_in[1];
    const float* W1    = (const float*)d_in[2];
    const float* a_s1  = (const float*)d_in[3];
    const float* a_d1  = (const float*)d_in[4];
    const float* b1    = (const float*)d_in[5];
    const float* W2    = (const float*)d_in[6];
    const float* a_s2  = (const float*)d_in[7];
    const float* a_d2  = (const float*)d_in[8];
    const float* b2    = (const float*)d_in[9];
    float*       out   = (float*)d_out;

    const int F = 256, H1 = 8, C1 = 32, HC1 = 256, C2 = 32;
    const int Nn = in_sizes[0] / F;   // 50000
    const int E  = in_sizes[1] / 2;   // 800000
    const int* src = ei;
    const int* dst = ei + E;

    // ---- carve workspace (~177 MB) ----
    char* p = (char*)d_ws;
    auto carve = [&](size_t bytes) {
        char* q = p;
        p += (bytes + 511) & ~(size_t)511;
        return q;
    };
    _Float16* xh    = (_Float16*)carve((size_t)Nn * F * 2);   // x f16, reused as elu(h1) f16
    _Float16* w1t   = (_Float16*)carve((size_t)HC1 * F * 2);  // W1^T [256][256]
    _Float16* w2t   = (_Float16*)carve((size_t)C2 * HC1 * 2); // W2^T [32][256]
    float* h1       = (float*)carve((size_t)Nn * HC1 * 4);
    float* out1     = (float*)carve((size_t)Nn * HC1 * 4);
    float* as1      = (float*)carve((size_t)Nn * H1 * 4);
    float* ad1      = (float*)carve((size_t)Nn * H1 * 4);
    float* dmax1    = (float*)carve((size_t)Nn * H1 * 4);
    float* den1     = (float*)carve((size_t)Nn * H1 * 4);
    float* ew1      = (float*)carve((size_t)E * H1 * 4);
    float* h2       = (float*)carve((size_t)Nn * C2 * 4);
    float* out2     = (float*)carve((size_t)Nn * C2 * 4);
    float* as2      = (float*)carve((size_t)Nn * 4);
    float* ad2      = (float*)carve((size_t)Nn * 4);
    float* dmax2    = (float*)carve((size_t)Nn * 4);
    float* den2     = (float*)carve((size_t)Nn * 4);
    float* ew2      = (float*)carve((size_t)E * 4);
    (void)ws_size; (void)n_in; (void)out_size;

    const int TB = 256;
    auto blocks = [&](long long n) { return (unsigned)((n + TB - 1) / TB); };

    // ---- preprocessing: f16 operands + transposed weights ----
    cvt_f32_to_f16<<<blocks((long long)Nn * F), TB, 0, stream>>>(x, xh, (long long)Nn * F);
    transpose_to_f16<<<blocks(F * HC1), TB, 0, stream>>>(W1, w1t, F, HC1);
    transpose_to_f16<<<blocks(HC1 * C2), TB, 0, stream>>>(W2, w2t, HC1, C2);

    // ---- init accumulators (every call; harness does not re-poison) ----
    fill_f32<<<blocks((long long)Nn * HC1), TB, 0, stream>>>(out1, 0.0f, (long long)Nn * HC1);
    fill_f32<<<blocks((long long)Nn * H1), TB, 0, stream>>>(den1, 0.0f, (long long)Nn * H1);
    fill_f32<<<blocks((long long)Nn * H1), TB, 0, stream>>>(dmax1, -INFINITY, (long long)Nn * H1);
    fill_f32<<<blocks((long long)Nn * C2), TB, 0, stream>>>(out2, 0.0f, (long long)Nn * C2);
    fill_f32<<<blocks(Nn), TB, 0, stream>>>(den2, 0.0f, Nn);
    fill_f32<<<blocks(Nn), TB, 0, stream>>>(dmax2, -INFINITY, Nn);

    const int Mt = Nn / 16;                       // 3125 (exact)
    // ---- layer 1: h1 = x @ W1  (WMMA, 16x64 per wave) ----
    {
        dim3 g((Mt + 3) / 4, HC1 / 64);
        gemm_f16_wmma<4><<<g, 128, 0, stream>>>(xh, w1t, h1, Nn, HC1, F);
    }
    alpha_kernel<<<blocks((long long)Nn * H1), TB, 0, stream>>>(h1, a_s1, a_d1, as1, ad1, Nn, H1, C1);
    edge_max_kernel<<<blocks((long long)E * H1), TB, 0, stream>>>(src, dst, as1, ad1, dmax1, E, H1);
    edge_exp_kernel<<<blocks((long long)E * H1), TB, 0, stream>>>(src, dst, as1, ad1, dmax1, ew1, den1, E, H1);
    edge_agg_kernel<<<blocks((long long)E * H1), TB, 0, stream>>>(src, dst, ew1, den1, h1, out1, E, H1, C1);
    finish1_kernel<<<blocks((long long)Nn * HC1), TB, 0, stream>>>(out1, b1, xh, (long long)Nn * HC1, HC1);

    // ---- layer 2: h2 = elu(h1) @ W2  (WMMA, 16x32 per wave) ----
    {
        dim3 g((Mt + 3) / 4, 1);
        gemm_f16_wmma<2><<<g, 128, 0, stream>>>(xh, w2t, h2, Nn, C2, HC1);
    }
    alpha_kernel<<<blocks(Nn), TB, 0, stream>>>(h2, a_s2, a_d2, as2, ad2, Nn, 1, C2);
    edge_max_kernel<<<blocks(E), TB, 0, stream>>>(src, dst, as2, ad2, dmax2, E, 1);
    edge_exp_kernel<<<blocks(E), TB, 0, stream>>>(src, dst, as2, ad2, dmax2, ew2, den2, E, 1);
    edge_agg_kernel<<<blocks(E), TB, 0, stream>>>(src, dst, ew2, den2, h2, out2, E, 1, C2);
    finish2_kernel<<<blocks(Nn), TB, 0, stream>>>(out2, b2, out, Nn, C2);
}
